// CoarseCurvaturePredictor_74689481277743
// MI455X (gfx1250) — matrix operand order repaired and checked by reading, not compiled, not run on hardware
//
#include <hip/hip_runtime.h>
#include <stdint.h>

// Problem constants (from reference: B=1, H=8, N=65536, D=128, BLOCK=64)
#define NHEAD   8
#define NTOK    65536
#define DDIM    128
#define BLKSZ   64
#define MBLK    1024                 // NTOK / BLKSZ
#define K_KEEP  52                   // ceil(0.05 * 1024)
#define AFF_SCALE 0.08838834764831845f  // 1/sqrt(128)

typedef __bf16 bf16;
typedef __attribute__((ext_vector_type(16))) bf16  v16bf;
typedef __attribute__((ext_vector_type(8)))  bf16  v8bf;
typedef __attribute__((ext_vector_type(8)))  float v8f;

// ---- WMMA fragment helpers (ISA 7.12.2 16-bit layouts, wave32) ----------
// A (16x32): K = k0 + 16*(lane/16) + e (contiguous, matches B)
static __device__ __forceinline__ v16bf load_a_frag(const bf16* __restrict__ row,
                                                    int k0, int hi) {
    return *reinterpret_cast<const v16bf*>(row + k0 + hi * 16);
}
// B (32x16): lane n = lane%16 is the column; K = k0 + 16*(lane/16) + e (contiguous)
static __device__ __forceinline__ v16bf load_b_frag(const bf16* __restrict__ row,
                                                    int k0, int hi) {
    return *reinterpret_cast<const v16bf*>(row + k0 + hi * 16);
}
static __device__ __forceinline__ v8f wmma_bf16(const v16bf& a, const v16bf& b,
                                                const v8f& c) {
    return __builtin_amdgcn_wmma_f32_16x16x32_bf16(false, a, false, b,
                                                   (short)0, c, false, false);
}

// -------------------------------------------------------------------------
// 1) Coarsen: per (head, block of 64 tokens) pick max-L2 token, emit bf16.
//    Fully coalesced float4 reads; wave-level shuffle reductions (wave32).
//    This is the HBM-roofline stage: 512 MB of q+k -> ~22us at 23.3 TB/s.
// -------------------------------------------------------------------------
__global__ __launch_bounds__(256) void coarsen_kernel(
    const float* __restrict__ x, bf16* __restrict__ out) {
    __shared__ float norms[BLKSZ];
    __shared__ int best;
    const int bid  = blockIdx.x;           // h*1024 + blk
    const int h    = bid >> 10, blk = bid & 1023;
    const int tid  = threadIdx.x;
    const int wave = tid >> 5, lane = tid & 31;
    const float* base = x + ((size_t)h * NTOK + (size_t)blk * BLKSZ) * DDIM;

    for (int pass = 0; pass < 8; ++pass) {
        const int t = pass * 8 + wave;     // 8 waves x 8 passes = 64 tokens
        const float4 f = reinterpret_cast<const float4*>(base + (size_t)t * DDIM)[lane];
        float s = f.x*f.x + f.y*f.y + f.z*f.z + f.w*f.w;
        for (int off = 16; off; off >>= 1) s += __shfl_xor(s, off);
        if (lane == 0) norms[t] = s;
    }
    __syncthreads();
    if (wave == 0) {
        float v = norms[lane]; int idx = lane;
        const float v2 = norms[lane + 32];
        if (v2 > v) { v = v2; idx = lane + 32; }
        for (int off = 16; off; off >>= 1) {
            const float ov = __shfl_xor(v, off);
            const int   oi = __shfl_xor(idx, off);
            if (ov > v || (ov == v && oi < idx)) { v = ov; idx = oi; }
        }
        if (lane == 0) best = idx;
    }
    __syncthreads();
    if (tid < DDIM)
        out[((size_t)h * MBLK + blk) * DDIM + tid] = (bf16)base[(size_t)best * DDIM + tid];
}

// -------------------------------------------------------------------------
// 2) Affinity: A = relu(qc kc^T / sqrt(D)). Each wave owns a 32x32 region
//    (2x2 WMMA tiles): A-frags reused across 2 N-tiles, B-frags across
//    2 M-tiles. Epilogue writes bf16 A row-major (Abf) and transposed
//    (AbfT) so the A@A GEMM gets contiguous B-side loads.
// -------------------------------------------------------------------------
static __device__ __forceinline__ void store_aff_tile(
    bf16* __restrict__ Abf, bf16* __restrict__ AbfT, size_t hb,
    int m0, int n0, int hi, int lo16, const v8f& c) {
    const int n = n0 + lo16;
    for (int r = 0; r < 8; ++r) {
        const int m = m0 + r + 8 * hi;         // C: VGPR r -> M = r + 8*(lane/16)
        float v = c[r] * AFF_SCALE;
        v = v > 0.f ? v : 0.f;                 // relu
        const bf16 bv = (bf16)v;               // native cvt on gfx1250
        Abf [hb + (size_t)m * MBLK + n] = bv;
        AbfT[hb + (size_t)n * MBLK + m] = bv;
    }
}

__global__ __launch_bounds__(256) void affinity_kernel(
    const bf16* __restrict__ qc, const bf16* __restrict__ kc,
    bf16* __restrict__ Abf, bf16* __restrict__ AbfT) {
    const int tid  = threadIdx.x;
    const int lane = tid & 31;
    const int gw   = blockIdx.x * 8 + (tid >> 5);
    const int h    = gw >> 10;                 // 1024 (32x32)-regions per head
    const int job  = gw & 1023;
    const int rm = job >> 5, rn = job & 31;
    const int hi = lane >> 4, lo16 = lane & 15;

    const bf16* q0 = qc + ((size_t)h * MBLK + (size_t)(rm * 32 + lo16)) * DDIM;
    const bf16* q1 = q0 + 16 * DDIM;
    const bf16* p0 = kc + ((size_t)h * MBLK + (size_t)(rn * 32 + lo16)) * DDIM;
    const bf16* p1 = p0 + 16 * DDIM;

    v8f c00 = {}, c01 = {}, c10 = {}, c11 = {};
    for (int k0 = 0; k0 < DDIM; k0 += 32) {
        const v16bf a0 = load_a_frag(q0, k0, hi);
        const v16bf a1 = load_a_frag(q1, k0, hi);
        const v16bf b0 = load_b_frag(p0, k0, hi);
        const v16bf b1 = load_b_frag(p1, k0, hi);
        c00 = wmma_bf16(a0, b0, c00);
        c01 = wmma_bf16(a0, b1, c01);
        c10 = wmma_bf16(a1, b0, c10);
        c11 = wmma_bf16(a1, b1, c11);
    }
    const size_t hb = (size_t)h * MBLK * MBLK;
    const int m0 = rm * 32, n0 = rn * 32;
    store_aff_tile(Abf, AbfT, hb, m0,      n0,      hi, lo16, c00);
    store_aff_tile(Abf, AbfT, hb, m0,      n0 + 16, hi, lo16, c01);
    store_aff_tile(Abf, AbfT, hb, m0 + 16, n0,      hi, lo16, c10);
    store_aff_tile(Abf, AbfT, hb, m0 + 16, n0 + 16, hi, lo16, c11);
}

// -------------------------------------------------------------------------
// 3) Degrees: deg_out = row sums of A (from Abf rows),
//             deg_in  = col sums of A (= row sums of AbfT). blockIdx.y picks.
// -------------------------------------------------------------------------
__global__ __launch_bounds__(256) void deg_kernel(
    const bf16* __restrict__ Abf, const bf16* __restrict__ AbfT,
    float* __restrict__ deg_out, float* __restrict__ deg_in) {
    const int row = blockIdx.x;                        // h*M + m
    const bf16* src = (blockIdx.y == 0 ? Abf : AbfT) + (size_t)row * MBLK;
    float s = 0.f;
    for (int j = threadIdx.x; j < MBLK; j += 256) s += (float)src[j];
    __shared__ float red[8];
    const int lane = threadIdx.x & 31, wave = threadIdx.x >> 5;
    for (int off = 16; off; off >>= 1) s += __shfl_xor(s, off);
    if (lane == 0) red[wave] = s;
    __syncthreads();
    if (threadIdx.x == 0) {
        float t = 0.f;
        for (int w = 0; w < 8; ++w) t += red[w];
        (blockIdx.y == 0 ? deg_out : deg_in)[row] = t;
    }
}

// -------------------------------------------------------------------------
// 4) Triangles + Forman-Ricci fused: T = A@A, K=1024, 2x2-tile register
//    blocking (128 WMMAs / wave). Operands (32 MB) are L2-resident; blocking
//    halves L2 traffic vs 1x1. Epilogue emits neg_frc directly.
// -------------------------------------------------------------------------
static __device__ __forceinline__ void store_frc_tile(
    float* __restrict__ negfrc, const float* __restrict__ deg_out_h,
    const float* __restrict__ deg_in_h, size_t hb,
    int m0, int n0, int hi, int lo16, const v8f& c) {
    const int n = n0 + lo16;
    const float din = deg_in_h[n];
    for (int r = 0; r < 8; ++r) {
        const int m = m0 + r + 8 * hi;
        // neg_frc = -(4 - deg_out - deg_in + 0.5*T)
        negfrc[hb + (size_t)m * MBLK + n] = deg_out_h[m] + din - 4.0f - 0.5f * c[r];
    }
}

__global__ __launch_bounds__(256) void tri_frc_kernel(
    const bf16* __restrict__ Abf, const bf16* __restrict__ AbfT,
    const float* __restrict__ deg_out, const float* __restrict__ deg_in,
    float* __restrict__ negfrc) {
    const int tid  = threadIdx.x;
    const int lane = tid & 31;
    const int gw   = blockIdx.x * 8 + (tid >> 5);
    const int h    = gw >> 10;
    const int job  = gw & 1023;
    const int rm = job >> 5, rn = job & 31;
    const int hi = lane >> 4, lo16 = lane & 15;

    const size_t hb = (size_t)h * MBLK * MBLK;
    const bf16* ar0 = Abf  + hb + (size_t)(rm * 32 + lo16) * MBLK;
    const bf16* ar1 = ar0 + 16 * MBLK;
    const bf16* br0 = AbfT + hb + (size_t)(rn * 32 + lo16) * MBLK;
    const bf16* br1 = br0 + 16 * MBLK;

    v8f c00 = {}, c01 = {}, c10 = {}, c11 = {};
    for (int k0 = 0; k0 < MBLK; k0 += 32) {
        if (k0 + 64 < MBLK) {                      // gfx1250 global_prefetch_b8
            __builtin_prefetch(ar0 + k0 + 64, 0, 0);
            __builtin_prefetch(br0 + k0 + 64, 0, 0);
        }
        const v16bf a0 = load_a_frag(ar0, k0, hi);
        const v16bf a1 = load_a_frag(ar1, k0, hi);
        const v16bf b0 = load_b_frag(br0, k0, hi);
        const v16bf b1 = load_b_frag(br1, k0, hi);
        c00 = wmma_bf16(a0, b0, c00);
        c01 = wmma_bf16(a0, b1, c01);
        c10 = wmma_bf16(a1, b0, c10);
        c11 = wmma_bf16(a1, b1, c11);
    }
    const float* doh = deg_out + (size_t)h * MBLK;
    const float* dih = deg_in  + (size_t)h * MBLK;
    const int m0 = rm * 32, n0 = rn * 32;
    store_frc_tile(negfrc, doh, dih, hb, m0,      n0,      hi, lo16, c00);
    store_frc_tile(negfrc, doh, dih, hb, m0,      n0 + 16, hi, lo16, c01);
    store_frc_tile(negfrc, doh, dih, hb, m0 + 16, n0,      hi, lo16, c10);
    store_frc_tile(negfrc, doh, dih, hb, m0 + 16, n0 + 16, hi, lo16, c11);
}

// -------------------------------------------------------------------------
// 5) Per-row top-k (k=52) threshold via binary search on order-preserving
//    uint keys in LDS; mask = neg_frc >= kth (ties kept, matches reference)
//    OR diagonal. Output as float 0/1.
// -------------------------------------------------------------------------
__global__ __launch_bounds__(256) void topk_mask_kernel(
    const float* __restrict__ negfrc, float* __restrict__ out) {
    __shared__ uint32_t keys[MBLK];
    __shared__ int cnt;
    const int rowg = blockIdx.x;                 // h*M + m
    const int m = rowg & (MBLK - 1);
    const float* src = negfrc + (size_t)rowg * MBLK;

    for (int j = threadIdx.x; j < MBLK; j += 256) {
        uint32_t u = __float_as_uint(src[j]);
        u = (u & 0x80000000u) ? ~u : (u | 0x80000000u);   // monotonic key
        keys[j] = u;
    }
    __syncthreads();

    uint32_t lo = 0u, hi = 0xFFFFFFFFu;   // invariant: count(>=lo) >= K_KEEP
    for (int it = 0; it < 32 && lo < hi; ++it) {
        const uint32_t mid = (uint32_t)(((uint64_t)lo + (uint64_t)hi + 1ull) >> 1);
        if (threadIdx.x == 0) cnt = 0;
        __syncthreads();
        int c = 0;
        for (int j = threadIdx.x; j < MBLK; j += 256) c += (keys[j] >= mid) ? 1 : 0;
        atomicAdd(&cnt, c);
        __syncthreads();
        const int total = cnt;
        __syncthreads();
        if (total >= K_KEEP) lo = mid; else hi = mid - 1;
    }
    for (int j = threadIdx.x; j < MBLK; j += 256) {
        const bool keep = (keys[j] >= lo) || (j == m);
        out[(size_t)rowg * MBLK + j] = keep ? 1.0f : 0.0f;
    }
}

// -------------------------------------------------------------------------
extern "C" void kernel_launch(void* const* d_in, const int* in_sizes, int n_in,
                              void* d_out, int out_size, void* d_ws, size_t ws_size,
                              hipStream_t stream) {
    const float* q = (const float*)d_in[0];
    const float* k = (const float*)d_in[1];
    float* out = (float*)d_out;

    char* ws = (char*)d_ws;
    // Workspace layout (~68 MB total):
    bf16*  qcbf    = (bf16*) (ws);                                 //  2 MB
    bf16*  kcbf    = (bf16*) (ws + (size_t)(2u  << 20));           //  2 MB
    bf16*  Abf     = (bf16*) (ws + (size_t)(4u  << 20));           // 16 MB
    bf16*  AbfT    = (bf16*) (ws + (size_t)(20u << 20));           // 16 MB
    float* negfrc  = (float*)(ws + (size_t)(36u << 20));           // 32 MB
    float* deg_out = (float*)(ws + (size_t)(68u << 20));           // 32 KB
    float* deg_in  = (float*)(ws + (size_t)(68u << 20) + (32u << 10));

    coarsen_kernel<<<NHEAD * MBLK, 256, 0, stream>>>(q, qcbf);
    coarsen_kernel<<<NHEAD * MBLK, 256, 0, stream>>>(k, kcbf);
    affinity_kernel<<<(NHEAD * 1024) / 8, 256, 0, stream>>>(qcbf, kcbf, Abf, AbfT);
    deg_kernel<<<dim3(NHEAD * MBLK, 2), 256, 0, stream>>>(Abf, AbfT, deg_out, deg_in);
    tri_frc_kernel<<<(NHEAD * 1024) / 8, 256, 0, stream>>>(Abf, AbfT, deg_out, deg_in, negfrc);
    topk_mask_kernel<<<NHEAD * MBLK, 256, 0, stream>>>(negfrc, out);
}
